// HybridQuantumCNN_65481071401497
// MI455X (gfx1250) — matrix operand order repaired and checked by reading, not compiled, not run on hardware
//
#include <hip/hip_runtime.h>
#include <stdint.h>

// ---------------------------------------------------------------------------
// HybridQuantumCNN fused kernel for MI455X (gfx1250, wave32, WMMA).
//
// Algebra: seq-len == 1  =>  softmax over single score == 1  =>  attention
// output == V.  Wq/bq/Wk/bk are dead.  Each layer:
//   seq += (LN1(seq) @ Wv + bv) @ Wo + bo
//   seq += gelu_tanh(LN2(seq) @ W1 + b1) @ W2 + b2
//
// One wave == 16 samples (WMMA M dim).  Matmuls on v_wmma_f32_16x16x32_f16
// (co-executes with VALU on CDNA5).  B-fragments zero-pad N>=10, so WMMA
// output columns 10..15 are exactly 0 -> all pad handling is branchless.
// GELU uses the CDNA5 hardware v_tanh_f32 transcendental.
// ---------------------------------------------------------------------------

typedef __attribute__((ext_vector_type(16))) _Float16 v16h;
typedef __attribute__((ext_vector_type(8)))  float    v8f;

#define PDIM   10
#define FDIM   64
#define NLAYER 2
#define WAVES  8   // 256 threads / 32
#define MAXBLK 1024

union FragU { uint4 q[2]; v16h h; };
union H2U  { _Float16 h[2]; uint32_t u; };

// Wave-local LDS ordering: LDS ops of one wave complete in order; the wait +
// scheduling barriers keep the compiler from reordering across the sync.
__device__ __forceinline__ void wave_lds_sync() {
  __builtin_amdgcn_wave_barrier();
  asm volatile("s_wait_dscnt 0" ::: "memory");
  __builtin_amdgcn_wave_barrier();
}

__device__ __forceinline__ v8f wmma16(v16h a, v16h b, v8f c) {
  return __builtin_amdgcn_wmma_f32_16x16x32_f16(
      false, a, false, b, (short)0, c, false, false);
}

// A-fragment (16x32 f16) from a row-major f16 LDS tile; two ds_load_b128.
__device__ __forceinline__ v16h load_a(const _Float16* base, int strideHalves,
                                       int kOff, int m, int hi) {
  const char* p = (const char*)base + ((m * strideHalves + kOff + hi * 8) << 1);
  FragU u;
  u.q[0] = *(const uint4*)(p);
  u.q[1] = *(const uint4*)(p + 32);
  return u.h;
}

// B-fragment: prebuilt per-lane dwords (8 per lane per tile) in LDS.
__device__ __forceinline__ v16h load_b(const uint32_t* wf, int t, int lane) {
  const uint4* p = (const uint4*)(wf + (t * 32 + lane) * 8);
  FragU u;
  u.q[0] = p[0];
  u.q[1] = p[1];
  return u.h;
}

// CDNA5 hardware transcendental tanh (TRANS32 unit).
__device__ __forceinline__ float fast_tanh(float v) {
  float r;
  asm volatile("v_tanh_f32 %0, %1" : "=v"(r) : "v"(v));
  return r;
}

__device__ __forceinline__ float gelu_tanh(float v) {
  // jax.nn.gelu default (approximate=True)
  float u = 0.7978845608028654f * (v + 0.044715f * v * v * v);
  return 0.5f * v * (1.0f + fast_tanh(u));
}

__global__ void __launch_bounds__(256)
proto_transformer_kernel(
    const float* __restrict__ x,    const float* __restrict__ protos,
    const float* __restrict__ Wv,   const float* __restrict__ bv,
    const float* __restrict__ Wo,   const float* __restrict__ bo,
    const float* __restrict__ ln1g, const float* __restrict__ ln1b,
    const float* __restrict__ ln2g, const float* __restrict__ ln2b,
    const float* __restrict__ W1,   const float* __restrict__ b1,
    const float* __restrict__ W2,   const float* __restrict__ b2,
    const float* __restrict__ Wc,   const float* __restrict__ bc,
    float* __restrict__ out, int B)
{
  // 16 B-fragment weight tiles (per layer: Wv, Wo, W1 n-tiles 0..3, W2 k-tiles 0..1)
  __shared__ uint32_t wfrag[16 * 32 * 8];            // 16 KB
  __shared__ float    pproto[PDIM][8];
  __shared__ float    pp2[PDIM];
  // zero-padded constant tables (index >= PDIM reads 0)
  __shared__ float    cLn1g[NLAYER][16], cLn1b[NLAYER][16];
  __shared__ float    cLn2g[NLAYER][16], cLn2b[NLAYER][16];
  __shared__ float    cBv[NLAYER][16],   cBo[NLAYER][16], cB2[NLAYER][16];
  __shared__ float    cB1[NLAYER][FDIM];
  __shared__ float    cWc[16];
  __shared__ float    cBcS;
  __shared__ float    seqS[WAVES][16][16];           // canonical f32 state
  __shared__ _Float16 hAS[WAVES][16][32];            // A input, K padded to 32
  __shared__ _Float16 gAS[WAVES][16][64];            // GELU activations (K=64)

  const int tid = threadIdx.x;

  // ---- stage prototypes + |p|^2 -----------------------------------------
  if (tid < PDIM * 8) pproto[tid >> 3][tid & 7] = protos[tid];
  __syncthreads();
  if (tid < PDIM) {
    float s = 0.f;
    #pragma unroll
    for (int d = 0; d < 8; ++d) { float p = pproto[tid][d]; s += p * p; }
    pp2[tid] = s;
  }
  // ---- stage zero-padded small constants --------------------------------
  for (int e = tid; e < NLAYER * 16; e += blockDim.x) {
    const int l = e >> 4, j = e & 15;
    const bool v = (j < PDIM);
    cLn1g[l][j] = v ? ln1g[l * PDIM + j] : 0.f;   // pad g=0,b=0 => LN pad cols == 0
    cLn1b[l][j] = v ? ln1b[l * PDIM + j] : 0.f;
    cLn2g[l][j] = v ? ln2g[l * PDIM + j] : 0.f;
    cLn2b[l][j] = v ? ln2b[l * PDIM + j] : 0.f;
    cBv[l][j]   = v ? bv[l * PDIM + j]   : 0.f;
    cBo[l][j]   = v ? bo[l * PDIM + j]   : 0.f;
    cB2[l][j]   = v ? b2[l * PDIM + j]   : 0.f;
  }
  for (int e = tid; e < NLAYER * FDIM; e += blockDim.x) cB1[e >> 6][e & 63] = b1[e];
  if (tid < 16) cWc[tid] = (tid < PDIM) ? Wc[tid] : 0.f;
  if (tid == 0) cBcS = bc[0];

  // ---- build f16 B-matrix fragments for all weight tiles ----------------
  for (int e = tid; e < 16 * 32; e += blockDim.x) {
    const int t = e >> 5, ln = e & 31;
    const int l = t >> 3, kind = t & 7;
    const int n = ln & 15, kb = (ln >> 4) * 8;
    #pragma unroll
    for (int r = 0; r < 8; ++r) {
      float f[2];
      #pragma unroll
      for (int s2 = 0; s2 < 2; ++s2) {
        const int k = kb + 2 * r + s2 + (r >= 4 ? 8 : 0);
        float v = 0.f;
        if (kind == 0)      { if (k < PDIM && n < PDIM) v = Wv[(l * PDIM + k) * PDIM + n]; }
        else if (kind == 1) { if (k < PDIM && n < PDIM) v = Wo[(l * PDIM + k) * PDIM + n]; }
        else if (kind <= 5) { if (k < PDIM)             v = W1[(l * PDIM + k) * FDIM + (kind - 2) * 16 + n]; }
        else { const int gk = (kind - 6) * 32 + k;
               if (gk < FDIM && n < PDIM)               v = W2[(l * FDIM + gk) * PDIM + n]; }
        f[s2] = v;
      }
      H2U pk; pk.h[0] = (_Float16)f[0]; pk.h[1] = (_Float16)f[1];
      wfrag[e * 8 + r] = pk.u;
    }
  }
  __syncthreads();

  const int wave = tid >> 5;
  const int lane = tid & 31;
  const int m    = lane & 15;   // A-matrix row / C-matrix column owned by lane
  const int hi   = lane >> 4;

  float*     seq = &seqS[wave][0][0];
  _Float16*  hA  = &hAS[wave][0][0];
  _Float16*  gA  = &gAS[wave][0][0];

  // per-lane, tile-invariant column biases (pad columns read 0)
  float rBv[NLAYER], rBo[NLAYER], rB2[NLAYER], rB1[NLAYER][4];
  #pragma unroll
  for (int l = 0; l < NLAYER; ++l) {
    rBv[l] = cBv[l][m]; rBo[l] = cBo[l][m]; rB2[l] = cB2[l][m];
    #pragma unroll
    for (int nt = 0; nt < 4; ++nt) rB1[l][nt] = cB1[l][nt * 16 + m];
  }

  const int numTiles   = (B + 15) >> 4;
  const int tileStride = gridDim.x * WAVES;

  for (int tile = blockIdx.x * WAVES + wave; tile < numTiles; tile += tileStride) {
    // prefetch next tile's x rows while this tile computes
    if (tile + tileStride < numTiles)
      __builtin_prefetch(x + ((size_t)(tile + tileStride) * 16 + m) * 8, 0, 3);

    // ---- RBF features: seq[m][j] = exp(-||x_m - p_j||^2), 5 j's per lane --
    {
      int s = tile * 16 + m; if (s > B - 1) s = B - 1;   // clamp tail, EXEC stays full
      const float4* xp = (const float4*)(x + (size_t)s * 8);
      const float4 xa = xp[0], xb = xp[1];
      const float x2 = xa.x*xa.x + xa.y*xa.y + xa.z*xa.z + xa.w*xa.w +
                       xb.x*xb.x + xb.y*xb.y + xb.z*xb.z + xb.w*xb.w;
      const int j0 = hi * 5;
      #pragma unroll
      for (int jj = 0; jj < 5; ++jj) {
        const int j = j0 + jj;
        const float* pr = pproto[j];
        float dp = xa.x*pr[0] + xa.y*pr[1] + xa.z*pr[2] + xa.w*pr[3]
                 + xb.x*pr[4] + xb.y*pr[5] + xb.z*pr[6] + xb.w*pr[7];
        seq[m * 16 + j] = __expf(-(x2 + pp2[j] - 2.f * dp));
      }
      // zero seq pad cols 10..15 (read as 0-producing inputs by padded LN)
      #pragma unroll
      for (int k = 0; k < 3; ++k) seq[m * 16 + 10 + hi * 3 + k] = 0.f;
      // zero this lane's half of hA row m: K = 10..31 must stay 0
      uint4 z; z.x = z.y = z.z = z.w = 0u;
      *(uint4*)&hA[m * 32 + hi * 16]     = z;
      *(uint4*)&hA[m * 32 + hi * 16 + 8] = z;
    }
    wave_lds_sync();

    #pragma unroll
    for (int l = 0; l < NLAYER; ++l) {
      // ---- LN1(row m) -> hA[m][0..11] as f16 (pairs split across halves) --
      {
        float mu = 0.f;
        #pragma unroll
        for (int j = 0; j < PDIM; ++j) mu += seq[m * 16 + j];
        mu *= (1.f / PDIM);
        float var = 0.f;
        #pragma unroll
        for (int j = 0; j < PDIM; ++j) { float d = seq[m * 16 + j] - mu; var += d * d; }
        const float rs = rsqrtf(var * (1.f / PDIM) + 1e-5f);
        #pragma unroll
        for (int k = 0; k < 3; ++k) {
          const int jp = hi * 3 + k;              // 0..5; jp==5 writes pad zeros
          const int j = 2 * jp;
          H2U pk;
          pk.h[0] = (_Float16)((seq[m*16+j]   - mu) * rs * cLn1g[l][j]   + cLn1b[l][j]);
          pk.h[1] = (_Float16)((seq[m*16+j+1] - mu) * rs * cLn1g[l][j+1] + cLn1b[l][j+1]);
          ((uint32_t*)hA)[m * 16 + jp] = pk.u;
        }
      }
      wave_lds_sync();

      // ---- v = LN1 @ Wv + bv (attention out == v for seq-len 1) ----------
      {
        v16h a = load_a(hA, 32, 0, m, hi);
        v8f c = {};
        c = wmma16(a, load_b(wfrag, l * 8 + 0, lane), c);
        // cols >= 10: c[i] == 0 (zero-padded B) and rBv == 0 -> stores 0: pad kept
        #pragma unroll
        for (int i = 0; i < 8; ++i)
          hA[(i + 8 * hi) * 32 + m] = (_Float16)(c[i] + rBv[l]);   // C->A relayout
      }
      wave_lds_sync();

      // ---- seq += v @ Wo + bo --------------------------------------------
      {
        v16h a = load_a(hA, 32, 0, m, hi);
        v8f c = {};
        c = wmma16(a, load_b(wfrag, l * 8 + 1, lane), c);
        #pragma unroll
        for (int i = 0; i < 8; ++i)
          seq[(i + 8 * hi) * 16 + m] += c[i] + rBo[l];             // pad cols += 0
      }
      wave_lds_sync();

      // ---- LN2(row m) -> hA[m][0..11] ------------------------------------
      {
        float mu = 0.f;
        #pragma unroll
        for (int j = 0; j < PDIM; ++j) mu += seq[m * 16 + j];
        mu *= (1.f / PDIM);
        float var = 0.f;
        #pragma unroll
        for (int j = 0; j < PDIM; ++j) { float d = seq[m * 16 + j] - mu; var += d * d; }
        const float rs = rsqrtf(var * (1.f / PDIM) + 1e-5f);
        #pragma unroll
        for (int k = 0; k < 3; ++k) {
          const int jp = hi * 3 + k;
          const int j = 2 * jp;
          H2U pk;
          pk.h[0] = (_Float16)((seq[m*16+j]   - mu) * rs * cLn2g[l][j]   + cLn2b[l][j]);
          pk.h[1] = (_Float16)((seq[m*16+j+1] - mu) * rs * cLn2g[l][j+1] + cLn2b[l][j+1]);
          ((uint32_t*)hA)[m * 16 + jp] = pk.u;
        }
      }
      wave_lds_sync();

      // ---- u = gelu(LN2 @ W1 + b1) -> gA (16 x 64 f16), 4 N-tiles --------
      {
        v16h a = load_a(hA, 32, 0, m, hi);    // reused for all 4 N-tiles
        #pragma unroll
        for (int nt = 0; nt < 4; ++nt) {
          v8f c = {};
          c = wmma16(a, load_b(wfrag, l * 8 + 2 + nt, lane), c);
          #pragma unroll
          for (int i = 0; i < 8; ++i)
            gA[(i + 8 * hi) * 64 + nt * 16 + m] = (_Float16)gelu_tanh(c[i] + rB1[l][nt]);
        }
      }
      wave_lds_sync();

      // ---- seq += u @ W2 + b2 (K=64 -> two chained K=32 WMMAs) -----------
      {
        v8f c = {};
        c = wmma16(load_a(gA, 64, 0,  m, hi), load_b(wfrag, l * 8 + 6, lane), c);
        c = wmma16(load_a(gA, 64, 32, m, hi), load_b(wfrag, l * 8 + 7, lane), c);
        #pragma unroll
        for (int i = 0; i < 8; ++i)
          seq[(i + 8 * hi) * 16 + m] += c[i] + rB2[l];             // pad cols += 0
      }
      wave_lds_sync();
    }

    // ---- classifier head + sigmoid ---------------------------------------
    if (hi == 0) {
      float acc = cBcS;
      #pragma unroll
      for (int j = 0; j < PDIM; ++j) acc += seq[m * 16 + j] * cWc[j];
      const int sidx = tile * 16 + m;
      if (sidx < B) out[sidx] = 1.f / (1.f + __expf(-acc));
    }
    wave_lds_sync();   // seq/hA reused by next tile
  }
}

extern "C" void kernel_launch(void* const* d_in, const int* in_sizes, int n_in,
                              void* d_out, int out_size, void* d_ws, size_t ws_size,
                              hipStream_t stream) {
  (void)n_in; (void)d_ws; (void)ws_size; (void)out_size;
  const float* x      = (const float*)d_in[0];
  const float* protos = (const float*)d_in[1];
  // d_in[2..5] = Wq, bq, Wk, bk: dead (seq-len 1 => softmax == 1 => o == v)
  const float* Wv   = (const float*)d_in[6];
  const float* bv   = (const float*)d_in[7];
  const float* Wo   = (const float*)d_in[8];
  const float* bo   = (const float*)d_in[9];
  const float* ln1g = (const float*)d_in[10];
  const float* ln1b = (const float*)d_in[11];
  const float* ln2g = (const float*)d_in[12];
  const float* ln2b = (const float*)d_in[13];
  const float* W1   = (const float*)d_in[14];
  const float* b1   = (const float*)d_in[15];
  const float* W2   = (const float*)d_in[16];
  const float* b2   = (const float*)d_in[17];
  const float* Wc   = (const float*)d_in[18];
  const float* bc   = (const float*)d_in[19];
  float* out = (float*)d_out;

  const int B = in_sizes[0] / 8;
  const int numTiles = (B + 15) / 16;
  int grid = (numTiles + WAVES - 1) / WAVES;
  if (grid > MAXBLK) grid = MAXBLK;    // persistent waves: amortize weight build

  proto_transformer_kernel<<<grid, 256, 0, stream>>>(
      x, protos, Wv, bv, Wo, bo, ln1g, ln1b, ln2g, ln2b,
      W1, b1, W2, b2, Wc, bc, out, B);
}
